// MLA_14035953123615
// MI455X (gfx1250) — compile-verified, matrix operand order checked by reading, not run on hardware
//
#include <hip/hip_runtime.h>
#include <cstdint>

#ifndef __has_builtin
#define __has_builtin(x) 0
#endif
#if defined(__HIP_DEVICE_COMPILE__) && \
    __has_builtin(__builtin_amdgcn_tensor_load_to_lds) && \
    __has_builtin(__builtin_amdgcn_s_wait_tensorcnt)
#define HAVE_TDM 1
#else
#define HAVE_TDM 0
#endif

// ---------------- CDNA5 WMMA types ----------------
typedef __attribute__((ext_vector_type(16))) __bf16 v16bf;
typedef __attribute__((ext_vector_type(8)))  float  v8f;

union FragA { v16bf v; unsigned u[8]; __bf16 e[16]; };
union FragC { v8f v; float f[8]; };

// 16-bit A/B fragment K-offset (ISA 7.12.2, 16x32 MxK):
// lane half hi: VGPR v<4 holds K = hi*8 + 2v, v>=4 holds K = 16 + hi*8 + 2(v-4)
__device__ __forceinline__ int frag_k0(int v, int hi) {
    return (v < 4) ? (hi * 8 + 2 * v) : (16 + hi * 8 + 2 * (v - 4));
}

#if HAVE_TDM
typedef __attribute__((ext_vector_type(4))) unsigned u32x4;
typedef __attribute__((ext_vector_type(8))) int      i32x8;
typedef __attribute__((ext_vector_type(4))) int      i32x4;

// Issue a TDM load of a 2D bf16 tile [tile_rows x tile_k] from a row-major
// tensor (row stride = tensor_d0 elements) into contiguous LDS at lds_off.
// D# layout per CDNA5 ISA 8.3/8.4 (group0: count/lds/global/type, group1:
// data_size=2B, dims/strides). Groups 2/3 zero (<=2D tensor).
__device__ __forceinline__ void tdm_load_2d(const __bf16* gptr, unsigned lds_off,
                                            int tile_k, int tile_rows,
                                            int tensor_d0, int tensor_d1) {
    unsigned long long ga = (unsigned long long)gptr;
    u32x4 g0 = { 1u,                                   // count=1, user mode
                 lds_off,                              // lds_addr (bytes)
                 (unsigned)(ga & 0xFFFFFFFFull),       // global_addr[31:0]
                 (unsigned)((ga >> 32) & 0x1FFFFFFull) | (2u << 30) }; // addr[56:32] | type=2
    unsigned ud0 = (unsigned)tensor_d0, ud1 = (unsigned)tensor_d1;
    i32x8 g1 = { (int)(1u << 16),                                  // data_size=1 (2 bytes)
                 (int)((ud0 & 0xFFFFu) << 16),                     // tensor_dim0[15:0]
                 (int)(((ud0 >> 16) & 0xFFFFu) | ((ud1 & 0xFFFFu) << 16)),
                 (int)(((ud1 >> 16) & 0xFFFFu) | ((unsigned)tile_k << 16)), // tile_dim0
                 (int)((unsigned)tile_rows & 0xFFFFu),             // tile_dim1 (tile_dim2=0)
                 (int)ud0,                                         // tensor_dim0_stride[31:0]
                 0, 0 };
    i32x4 z4 = { 0, 0, 0, 0 };
    i32x8 z8 = { 0, 0, 0, 0, 0, 0, 0, 0 };
    // 6-arg form (clang-23 / therock headers): (g0, g1, g2, g3, g4, cpol)
    __builtin_amdgcn_tensor_load_to_lds(g0, g1, z4, z4, z8, 0);
}
#endif

// ---------------- GEMM: C[M,N] = A[M,K] * Bt[N,K]^T, bf16 in, bf16/f32 out ----------------
#define BM 128
#define BN 128
#define BKK 64

template <typename OutT>
__global__ __launch_bounds__(256)
void gemm_bf16(const __bf16* __restrict__ A, const __bf16* __restrict__ Bt,
               OutT* __restrict__ C, int M, int N, int K) {
    __shared__ __bf16 lA[2][BM][BKK];   // [buf][m][k]
    __shared__ __bf16 lB[2][BN][BKK];   // [buf][n][k]

    const int tid  = threadIdx.x;
    const int lane = tid & 31;
    const int w    = tid >> 5;
    const int hi   = lane >> 4;
    const int lo   = lane & 15;
    const int wm   = w >> 2;
    const int wn   = w & 3;
    const int gm0  = blockIdx.y * BM;
    const int gn0  = blockIdx.x * BN;
    const int rm   = wm * 64;
    const int rn   = wn * 32;

    FragC acc[4][2];
#pragma unroll
    for (int mt = 0; mt < 4; ++mt)
#pragma unroll
        for (int nt = 0; nt < 2; ++nt)
#pragma unroll
            for (int i = 0; i < 8; ++i) acc[mt][nt].f[i] = 0.f;

    const int nk = K / BKK;

#if HAVE_TDM
    // wave 0 drives the Tensor Data Mover; double-buffered prefetch
    if (tid < 32) {
        tdm_load_2d(A  + (size_t)gm0 * K, (unsigned)(uintptr_t)&lA[0][0][0], BKK, BM, K, M);
        tdm_load_2d(Bt + (size_t)gn0 * K, (unsigned)(uintptr_t)&lB[0][0][0], BKK, BN, K, N);
    }
#endif

    for (int it = 0; it < nk; ++it) {
        const int cur = it & 1;
#if HAVE_TDM
        if (tid < 32) __builtin_amdgcn_s_wait_tensorcnt(0);
        __syncthreads();                    // tile[cur] ready; everyone past tile[1-cur]
        if (it + 1 < nk && tid < 32) {
            const int k1 = (it + 1) * BKK;
            tdm_load_2d(A  + (size_t)gm0 * K + k1,
                        (unsigned)(uintptr_t)&lA[1 - cur][0][0], BKK, BM, K, M);
            tdm_load_2d(Bt + (size_t)gn0 * K + k1,
                        (unsigned)(uintptr_t)&lB[1 - cur][0][0], BKK, BN, K, N);
        }
#else
        const int k0 = it * BKK;
        __syncthreads();
#pragma unroll
        for (int i = 0; i < 4; ++i) {       // 1024 8-elem chunks per tile
            int chunk = i * 256 + tid;
            int r  = chunk >> 3;
            int c8 = (chunk & 7) * 8;
            *(uint4*)&lA[cur][r][c8] = *(const uint4*)&A[(size_t)(gm0 + r) * K + k0 + c8];
            *(uint4*)&lB[cur][r][c8] = *(const uint4*)&Bt[(size_t)(gn0 + r) * K + k0 + c8];
        }
        __syncthreads();
#endif
        const __bf16 (*cA)[BKK] = lA[cur];
        const __bf16 (*cB)[BKK] = lB[cur];
#pragma unroll
        for (int kc = 0; kc < 2; ++kc) {
            FragA bfr[2];
#pragma unroll
            for (int nt = 0; nt < 2; ++nt)
#pragma unroll
                for (int v = 0; v < 8; ++v)
                    bfr[nt].u[v] = *reinterpret_cast<const unsigned*>(
                        &cB[rn + nt * 16 + lo][kc * 32 + frag_k0(v, hi)]);
#pragma unroll
            for (int mt = 0; mt < 4; ++mt) {
                FragA afr;
#pragma unroll
                for (int v = 0; v < 8; ++v)
                    afr.u[v] = *reinterpret_cast<const unsigned*>(
                        &cA[rm + mt * 16 + lo][kc * 32 + frag_k0(v, hi)]);
#pragma unroll
                for (int nt = 0; nt < 2; ++nt)
                    acc[mt][nt].v = __builtin_amdgcn_wmma_f32_16x16x32_bf16(
                        false, afr.v, false, bfr[nt].v, (short)0, acc[mt][nt].v, false, false);
            }
        }
        __syncthreads();
    }

#pragma unroll
    for (int mt = 0; mt < 4; ++mt)
#pragma unroll
        for (int nt = 0; nt < 2; ++nt)
#pragma unroll
            for (int i = 0; i < 8; ++i) {
                int row = gm0 + rm + mt * 16 + hi * 8 + i;
                int col = gn0 + rn + nt * 16 + lo;
                C[(size_t)row * N + col] = (OutT)acc[mt][nt].f[i];
            }
}

// ---------------- f32 -> bf16 conversion (vectorized x4) ----------------
__global__ void cvt_bf16(const float* __restrict__ in, __bf16* __restrict__ out, int n4) {
    int i = blockIdx.x * blockDim.x + threadIdx.x;
    if (i >= n4) return;
    float4 f = *reinterpret_cast<const float4*>(in + (size_t)i * 4);
    __bf16* o = out + (size_t)i * 4;
    o[0] = (__bf16)f.x; o[1] = (__bf16)f.y; o[2] = (__bf16)f.z; o[3] = (__bf16)f.w;
}

// ---------------- f32 [K,N] -> bf16 [N,K] transpose ----------------
__global__ __launch_bounds__(256)
void transpose_bf16(const float* __restrict__ W, __bf16* __restrict__ Wt, int K, int N) {
    __shared__ float tile[32][33];
    int bn = blockIdx.x * 32, bk = blockIdx.y * 32;
    int tx = threadIdx.x & 31, ty = threadIdx.x >> 5;   // 32 x 8
#pragma unroll
    for (int i = 0; i < 32; i += 8)
        tile[ty + i][tx] = W[(size_t)(bk + ty + i) * N + bn + tx];
    __syncthreads();
#pragma unroll
    for (int i = 0; i < 32; i += 8)
        Wt[(size_t)(bn + ty + i) * K + bk + tx] = (__bf16)tile[tx][ty + i];
}

// ---------------- RoPE (in-place on bf16 [B*T, H*64]) ----------------
__global__ void rope_bf16(__bf16* __restrict__ p, int BT, int T) {
    int idx = blockIdx.x * blockDim.x + threadIdx.x;
    if (idx >= BT * 16 * 32) return;
    int i  = idx & 31;
    int h  = (idx >> 5) & 15;
    int bt = idx >> 9;
    int t  = bt % T;
    float inv = __expf(-((float)(2 * i) / 64.f) * 9.210340371976184f); // 10000^(-2i/64)
    float ang = (float)t * inv;
    float s, c;
    __sincosf(ang, &s, &c);
    __bf16* base = p + (size_t)bt * (16 * 64) + h * 64;
    float x1 = (float)base[i], x2 = (float)base[i + 32];
    base[i]      = (__bf16)(x1 * c - x2 * s);
    base[i + 32] = (__bf16)(x1 * s + x2 * c);
}

// ---------------- Flash attention with WMMA (16 q-rows per wave) ----------------
__global__ __launch_bounds__(256)
void mla_attention(const __bf16* __restrict__ qc, const __bf16* __restrict__ qp,
                   const __bf16* __restrict__ kc, const __bf16* __restrict__ kp,
                   const __bf16* __restrict__ V,  __bf16* __restrict__ O) {
    const int T_ = 2048, Hn = 16;
    int bid = blockIdx.x;             // B*H*(T/128) = 512
    int qt = bid & 15;
    int h  = (bid >> 4) & 15;
    int b  = bid >> 8;

    const int tid  = threadIdx.x;
    const int lane = tid & 31;
    const int w    = tid >> 5;
    const int hi   = lane >> 4;
    const int lo   = lane & 15;
    const int qt0  = qt * 128;
    const int q0w  = qt0 + w * 16;

    __shared__ __bf16 lK[32][128];    // [key][d]
    __shared__ __bf16 lV[128][32];    // [d][key] (transposed for B-frags)
    __shared__ __bf16 lP[8][16][32];  // per-wave P transpose staging

    // Q fragments: raw 32-bit copies of bf16 pairs; 1/sqrt(HD) folded in later
    FragA qf[4];
    {
        size_t rowbase = (size_t)b * T_ + q0w + lo;
        const __bf16* qcrow = qc + rowbase * 1024 + h * 64;
        const __bf16* qprow = qp + rowbase * 1024 + h * 64;
#pragma unroll
        for (int c4 = 0; c4 < 4; ++c4)
#pragma unroll
            for (int v = 0; v < 8; ++v) {
                int d0 = c4 * 32 + frag_k0(v, hi);   // even; pair stays on one side of 64
                const __bf16* src = (d0 < 64) ? (qcrow + d0) : (qprow + (d0 - 64));
                qf[c4].u[v] = *reinterpret_cast<const unsigned*>(src);
            }
    }

    const float scale = 0.08838834764831845f; // 1/sqrt(128)
    float mrow[8], lrow[8];
#pragma unroll
    for (int i = 0; i < 8; ++i) { mrow[i] = -1e30f; lrow[i] = 0.f; }
    FragC accO[8];
#pragma unroll
    for (int dt = 0; dt < 8; ++dt)
#pragma unroll
        for (int i = 0; i < 8; ++i) accO[dt].f[i] = 0.f;

    const int nkt = (qt0 + 128) / 32;
    for (int kt = 0; kt < nkt; ++kt) {
        int kt0 = kt * 32;
        __syncthreads();
        // K tile: uint4 copies (8 bf16), chunks never straddle content/rope split
#pragma unroll
        for (int i = 0; i < 2; ++i) {
            int chunk = i * 256 + tid;             // 512 chunks
            int kk = chunk >> 4;
            int c8 = (chunk & 15) * 8;
            size_t krow = (size_t)b * T_ + kt0 + kk;
            const __bf16* src = (c8 < 64) ? (kc + krow * 1024 + h * 64 + c8)
                                          : (kp + krow * 1024 + h * 64 + (c8 - 64));
            *(uint4*)&lK[kk][c8] = *(const uint4*)src;
        }
        // V tile transposed (per-element)
#pragma unroll
        for (int i = 0; i < 16; ++i) {
            int flat = i * 256 + tid;
            int kk = flat >> 7, dd = flat & 127;
            size_t krow = (size_t)b * T_ + kt0 + kk;
            lV[dd][kk] = V[krow * (Hn * 128) + h * 128 + dd];
        }
        __syncthreads();

        // S = Q @ K^T
        FragC accS[2];
#pragma unroll
        for (int nt = 0; nt < 2; ++nt) {
#pragma unroll
            for (int i = 0; i < 8; ++i) accS[nt].f[i] = 0.f;
#pragma unroll
            for (int c4 = 0; c4 < 4; ++c4) {
                FragA kf;
#pragma unroll
                for (int v = 0; v < 8; ++v)
                    kf.u[v] = *reinterpret_cast<const unsigned*>(
                        &lK[nt * 16 + lo][c4 * 32 + frag_k0(v, hi)]);
                accS[nt].v = __builtin_amdgcn_wmma_f32_16x16x32_bf16(
                    false, qf[c4].v, false, kf.v, (short)0, accS[nt].v, false, false);
            }
            int key = kt0 + nt * 16 + lo;
#pragma unroll
            for (int i = 0; i < 8; ++i) {
                float sv = accS[nt].f[i] * scale;
                accS[nt].f[i] = (key > q0w + hi * 8 + i) ? -1e30f : sv;
            }
        }

        // online softmax; rows live at (vgpr i, lane half), cols across 16 lanes
        float mt_[8];
#pragma unroll
        for (int i = 0; i < 8; ++i) mt_[i] = fmaxf(accS[0].f[i], accS[1].f[i]);
        for (int m = 1; m < 16; m <<= 1)
#pragma unroll
            for (int i = 0; i < 8; ++i) mt_[i] = fmaxf(mt_[i], __shfl_xor(mt_[i], m, 32));
        float alpha[8];
#pragma unroll
        for (int i = 0; i < 8; ++i) {
            float mn = fmaxf(mrow[i], mt_[i]);
            alpha[i] = __expf(mrow[i] - mn);
            mrow[i]  = mn;
        }
        float rsum[8];
#pragma unroll
        for (int i = 0; i < 8; ++i) rsum[i] = 0.f;
#pragma unroll
        for (int nt = 0; nt < 2; ++nt)
#pragma unroll
            for (int i = 0; i < 8; ++i) {
                float pv = __expf(accS[nt].f[i] - mrow[i]);
                accS[nt].f[i] = pv;
                rsum[i] += pv;
            }
        for (int m = 1; m < 16; m <<= 1)
#pragma unroll
            for (int i = 0; i < 8; ++i) rsum[i] += __shfl_xor(rsum[i], m, 32);
#pragma unroll
        for (int i = 0; i < 8; ++i) lrow[i] = lrow[i] * alpha[i] + rsum[i];
#pragma unroll
        for (int dt = 0; dt < 8; ++dt)
#pragma unroll
            for (int i = 0; i < 8; ++i) accO[dt].f[i] *= alpha[i];

        // transpose P through per-wave LDS
#pragma unroll
        for (int nt = 0; nt < 2; ++nt)
#pragma unroll
            for (int i = 0; i < 8; ++i)
                lP[w][hi * 8 + i][nt * 16 + lo] = (__bf16)accS[nt].f[i];
        __syncthreads();

        FragA pf;
#pragma unroll
        for (int v = 0; v < 8; ++v)
            pf.u[v] = *reinterpret_cast<const unsigned*>(&lP[w][lo][frag_k0(v, hi)]);
#pragma unroll
        for (int dt = 0; dt < 8; ++dt) {
            FragA vf;
#pragma unroll
            for (int v = 0; v < 8; ++v)
                vf.u[v] = *reinterpret_cast<const unsigned*>(
                    &lV[dt * 16 + lo][frag_k0(v, hi)]);
            accO[dt].v = __builtin_amdgcn_wmma_f32_16x16x32_bf16(
                false, pf.v, false, vf.v, (short)0, accO[dt].v, false, false);
        }
    }

#pragma unroll
    for (int dt = 0; dt < 8; ++dt)
#pragma unroll
        for (int i = 0; i < 8; ++i) {
            int trow = q0w + hi * 8 + i;
            O[((size_t)b * T_ + trow) * (Hn * 128) + h * 128 + dt * 16 + lo] =
                (__bf16)(accO[dt].f[i] / lrow[i]);
        }
}

// ---------------- Launcher ----------------
extern "C" void kernel_launch(void* const* d_in, const int* in_sizes, int n_in,
                              void* d_out, int out_size, void* d_ws, size_t ws_size,
                              hipStream_t stream) {
    (void)in_sizes; (void)n_in; (void)out_size; (void)ws_size;
    const int Bb = 2, Tt = 2048, DM = 2048, QR = 512, DL = 512;
    const int BT = Bb * Tt;                  // 4096

    const float* x        = (const float*)d_in[0];
    const float* Wq_down  = (const float*)d_in[1];
    const float* Wq_up    = (const float*)d_in[2];
    const float* Wq_rope  = (const float*)d_in[3];
    const float* Wkv_down = (const float*)d_in[4];
    const float* Wk_up    = (const float*)d_in[5];
    const float* Wv_up    = (const float*)d_in[6];
    const float* Wk_rope  = (const float*)d_in[7];
    const float* Wo       = (const float*)d_in[8];
    float* out = (float*)d_out;

    // workspace layout (bf16 elements)
    __bf16* ws = (__bf16*)d_ws;
    size_t o = 0;
    __bf16* xb     = ws + o; o += (size_t)BT * DM;
    __bf16* wqd_t  = ws + o; o += (size_t)QR * DM;      // [512,2048]
    __bf16* wqu_t  = ws + o; o += (size_t)1024 * QR;    // [1024,512]
    __bf16* wqr_t  = ws + o; o += (size_t)1024 * DM;    // [1024,2048]
    __bf16* wkvd_t = ws + o; o += (size_t)DL * DM;      // [512,2048]
    __bf16* wku_t  = ws + o; o += (size_t)1024 * DL;    // [1024,512]
    __bf16* wvu_t  = ws + o; o += (size_t)2048 * DL;    // [2048,512]
    __bf16* wkr_t  = ws + o; o += (size_t)1024 * DM;    // [1024,2048]
    __bf16* wo_t   = ws + o; o += (size_t)DM * 2048;    // [2048,2048]
    __bf16* qd     = ws + o; o += (size_t)BT * QR;
    __bf16* qcont  = ws + o; o += (size_t)BT * 1024;
    __bf16* qpos   = ws + o; o += (size_t)BT * 1024;
    __bf16* kvlat  = ws + o; o += (size_t)BT * DL;
    __bf16* kcont  = ws + o; o += (size_t)BT * 1024;
    __bf16* kpos   = ws + o; o += (size_t)BT * 1024;
    __bf16* vmat   = ws + o; o += (size_t)BT * 2048;
    __bf16* attn   = ws + o;

    dim3 blk(256);

    // convert activations, convert+transpose weights ([K,N] -> bf16 [N,K])
    {
        int n4 = (BT * DM) / 4;
        cvt_bf16<<<dim3((n4 + 255) / 256), blk, 0, stream>>>(x, xb, n4);
        auto tr = [&](const float* W, __bf16* Wt, int K, int N) {
            transpose_bf16<<<dim3(N / 32, K / 32), blk, 0, stream>>>(W, Wt, K, N);
        };
        tr(Wq_down,  wqd_t,  DM, QR);
        tr(Wq_up,    wqu_t,  QR, 1024);
        tr(Wq_rope,  wqr_t,  DM, 1024);
        tr(Wkv_down, wkvd_t, DM, DL);
        tr(Wk_up,    wku_t,  DL, 1024);
        tr(Wv_up,    wvu_t,  DL, 2048);
        tr(Wk_rope,  wkr_t,  DM, 1024);
        tr(Wo,       wo_t,   DM, 2048);
    }

    auto gemmb = [&](const __bf16* A, const __bf16* Bt, __bf16* C, int M, int N, int K) {
        gemm_bf16<__bf16><<<dim3(N / BN, M / BM), blk, 0, stream>>>(A, Bt, C, M, N, K);
    };

    gemmb(xb,    wqd_t,  qd,    BT, QR,   DM);
    gemmb(qd,    wqu_t,  qcont, BT, 1024, QR);
    gemmb(xb,    wqr_t,  qpos,  BT, 1024, DM);
    gemmb(xb,    wkvd_t, kvlat, BT, DL,   DM);
    gemmb(kvlat, wku_t,  kcont, BT, 1024, DL);
    gemmb(xb,    wkr_t,  kpos,  BT, 1024, DM);
    gemmb(kvlat, wvu_t,  vmat,  BT, 2048, DL);

    int ropeThreads = BT * 16 * 32;
    rope_bf16<<<dim3((ropeThreads + 255) / 256), blk, 0, stream>>>(qpos, BT, Tt);
    rope_bf16<<<dim3((ropeThreads + 255) / 256), blk, 0, stream>>>(kpos, BT, Tt);

    mla_attention<<<dim3(Bb * 16 * (Tt / 128)), blk, 0, stream>>>(
        qcont, qpos, kcont, kpos, vmat, attn);

    gemm_bf16<float><<<dim3(DM / BN, BT / BM), blk, 0, stream>>>(
        attn, wo_t, out, BT, DM, 2048);
}
